// create_fc_head_3702261809174
// MI455X (gfx1250) — compile-verified, matrix operand order checked by reading, not run on hardware
//
#include <hip/hip_runtime.h>

// ---------------------------------------------------------------------------
// Per-channel MLP head (3 layers, ReLU+BN between) for MI455X / gfx1250.
//  - GEMM: V_WMMA_F32_16X16X32_BF16, bf16x3 error-compensated split
//    (A_hi*B_hi + A_hi*B_lo + A_lo*B_hi), split done with v_perm_b32
//    (truncation split, ~2^-16 rel err, 3 VALU/float).
//  - BN folded into the NEXT layer's A-staging (a*scale+shift), so no
//    standalone elementwise BN pass touches HBM.
//  - Global loads software-pipelined one K-chunk ahead + global_prefetch_b8
//    two chunks ahead.
// d_ws: h1 (64MB) | h2 (64MB) | sum | sumsq | scale | shift (128KB each)
// ---------------------------------------------------------------------------

typedef __bf16        v16bf __attribute__((ext_vector_type(16)));
typedef float         v8f   __attribute__((ext_vector_type(8)));
typedef float         f32x4 __attribute__((ext_vector_type(4)));
typedef unsigned int  u32x4 __attribute__((ext_vector_type(4)));

#define N_BATCH 512
#define N_CHAN  32
#define L_FEAT  1024
#define BN_EPS  1e-5f
#define LDS_ROW 40   // padded row length in ushorts (80B, 16B-aligned rows)

// ---- bf16 split helpers ----------------------------------------------------
// hi = top 16 bits of f (truncation), lo = top 16 bits of (f - hi).
// Packs two results per u32 with one v_perm_b32.
__device__ __forceinline__ void cvt16_store(f32x4 a0, f32x4 a1, f32x4 a2, f32x4 a3,
                                            unsigned short* dhi, unsigned short* dlo) {
    f32x4 av[4] = {a0, a1, a2, a3};
    unsigned hp[8], lp[8];
#pragma unroll
    for (int i = 0; i < 8; ++i) {
        float f0 = av[i >> 1][(i & 1) * 2 + 0];
        float f1 = av[i >> 1][(i & 1) * 2 + 1];
        unsigned u0 = __builtin_bit_cast(unsigned, f0);
        unsigned u1 = __builtin_bit_cast(unsigned, f1);
        float h0f = __builtin_bit_cast(float, u0 & 0xffff0000u);
        float h1f = __builtin_bit_cast(float, u1 & 0xffff0000u);
        float l0  = f0 - h0f;
        float l1  = f1 - h1f;
        // result bytes: [f0.b2, f0.b3, f1.b2, f1.b3]  (src0=hi-source)
        hp[i] = __builtin_amdgcn_perm(u1, u0, 0x07060302u);
        lp[i] = __builtin_amdgcn_perm(__builtin_bit_cast(unsigned, l1),
                                      __builtin_bit_cast(unsigned, l0), 0x07060302u);
    }
    u32x4 h0v = {hp[0], hp[1], hp[2], hp[3]};
    u32x4 h1v = {hp[4], hp[5], hp[6], hp[7]};
    u32x4 l0v = {lp[0], lp[1], lp[2], lp[3]};
    u32x4 l1v = {lp[4], lp[5], lp[6], lp[7]};
    ((u32x4*)dhi)[0] = h0v;
    ((u32x4*)dhi)[1] = h1v;
    ((u32x4*)dlo)[0] = l0v;
    ((u32x4*)dlo)[1] = l1v;
}

// Load one 16x32 bf16 WMMA operand fragment from an LDS plane.
// ISA 16-bit operand layout: lanes 0-15 hold rows 0-15 with K 0-7 (VGPR0-3)
// and K 16-23 (VGPR4-7); lanes 16-31 hold K 8-15 and K 24-31.
__device__ __forceinline__ v16bf frag_ld(const unsigned short* plane, int rowbase, int lane) {
    int r    = rowbase + (lane & 15);
    int koff = (lane >> 4) << 3;  // 0 or 8
    const unsigned short* p = plane + r * LDS_ROW + koff;
    union U { v16bf v; u32x4 u[2]; } t;
    t.u[0] = *(const u32x4*)(p);
    t.u[1] = *(const u32x4*)(p + 16);
    return t.v;
}

struct Tile16 { f32x4 v[4]; };
__device__ __forceinline__ Tile16 ld16(const float* p) {
    Tile16 t;
    t.v[0] = ((const f32x4*)p)[0];
    t.v[1] = ((const f32x4*)p)[1];
    t.v[2] = ((const f32x4*)p)[2];
    t.v[3] = ((const f32x4*)p)[3];
    return t;
}

// ---------------------------------------------------------------------------
// GEMM: Out[c, n, m] = sum_l BN(A[c, n, l]) * W[c, m, l] + bias[c, m]
// BN(a) = a*aScale[c,l] + aShift[c,l] when aScale != nullptr.
// Block = 128(n) x 128(m) tile of one channel, 256 threads = 8 waves,
// wave grid 2(n) x 4(m): each wave owns 64(n) x 32(m) = 4x2 WMMA tiles.
// ---------------------------------------------------------------------------
__global__ __launch_bounds__(256)
void gemm_bf16x3(const float* __restrict__ A, long long aChan, int aRow,
                 const float* __restrict__ aScale, const float* __restrict__ aShift,
                 const float* __restrict__ Wl, long long wChan,
                 const float* __restrict__ bl, long long bChan,
                 float* __restrict__ Out, long long oChan, int oRow,
                 int doRelu, float* __restrict__ sSum, float* __restrict__ sSq) {
    const int c    = blockIdx.z;
    const int bm   = blockIdx.x;     // 0..7, 128 output features each
    const int bn   = blockIdx.y;     // 0..3, 128 batch rows each
    const int tid  = threadIdx.x;
    const int lane = tid & 31;
    const int w    = tid >> 5;       // 0..7
    const int wm   = w & 3;          // wave col: 32 m
    const int wn   = w >> 2;         // wave row: 64 n

    __shared__ unsigned short Ah[128 * LDS_ROW];
    __shared__ unsigned short Al[128 * LDS_ROW];
    __shared__ unsigned short Bh[128 * LDS_ROW];
    __shared__ unsigned short Bl[128 * LDS_ROW];

    const float* aBase = A  + (long long)c * aChan + (long long)(bn * 128) * aRow;
    const float* wBase = Wl + (long long)c * wChan + (long long)(bm * 128) * L_FEAT;

    v8f acc[4][2];
#pragma unroll
    for (int i = 0; i < 4; ++i)
#pragma unroll
        for (int j = 0; j < 2; ++j) {
            v8f z = {0.f, 0.f, 0.f, 0.f, 0.f, 0.f, 0.f, 0.f};
            acc[i][j] = z;
        }

    const int ldr = tid >> 1;        // row 0..127 staged by this thread
    const int ldk = (tid & 1) * 16;  // K segment 0 or 16
    const float* aRowPtr = aBase + (long long)ldr * aRow + ldk;
    const float* wRowPtr = wBase + (long long)ldr * L_FEAT + ldk;
    const float* scBase  = aScale ? (aScale + (c << 10) + ldk) : nullptr;
    const float* shBase  = aShift ? (aShift + (c << 10) + ldk) : nullptr;

    // software pipeline: chunk k0 held in regs while chunk k0-32 computes
    Tile16 aReg = ld16(aRowPtr);
    Tile16 wReg = ld16(wRowPtr);

    for (int k0 = 0; k0 < L_FEAT; k0 += 32) {
        // -- normalize A (folded BN) and split both tiles into LDS ---------
        f32x4 af[4];
        if (scBase) {
#pragma unroll
            for (int j = 0; j < 4; ++j) {
                f32x4 sc = ((const f32x4*)(scBase + k0))[j];
                f32x4 sh = ((const f32x4*)(shBase + k0))[j];
                af[j] = aReg.v[j] * sc + sh;
            }
        } else {
#pragma unroll
            for (int j = 0; j < 4; ++j) af[j] = aReg.v[j];
        }
        cvt16_store(af[0], af[1], af[2], af[3],
                    &Ah[ldr * LDS_ROW + ldk], &Al[ldr * LDS_ROW + ldk]);
        cvt16_store(wReg.v[0], wReg.v[1], wReg.v[2], wReg.v[3],
                    &Bh[ldr * LDS_ROW + ldk], &Bl[ldr * LDS_ROW + ldk]);
        __syncthreads();

        // -- issue next chunk's global loads early (overlap with WMMA) -----
        if (k0 + 32 < L_FEAT) {
            aReg = ld16(aRowPtr + k0 + 32);
            wReg = ld16(wRowPtr + k0 + 32);
            if (k0 + 64 < L_FEAT) {
                __builtin_prefetch(aRowPtr + k0 + 64, 0, 3);
                __builtin_prefetch(wRowPtr + k0 + 64, 0, 3);
            }
        }

        // -- fragments + WMMA ---------------------------------------------
        v16bf bhi[2], blo[2];
#pragma unroll
        for (int ms = 0; ms < 2; ++ms) {
            bhi[ms] = frag_ld(Bh, wm * 32 + ms * 16, lane);
            blo[ms] = frag_ld(Bl, wm * 32 + ms * 16, lane);
        }
#pragma unroll
        for (int ns = 0; ns < 4; ++ns) {
            v16bf ahi = frag_ld(Ah, wn * 64 + ns * 16, lane);
            v16bf alo = frag_ld(Al, wn * 64 + ns * 16, lane);
#pragma unroll
            for (int ms = 0; ms < 2; ++ms) {
                acc[ns][ms] = __builtin_amdgcn_wmma_f32_16x16x32_bf16(
                    false, ahi, false, bhi[ms], (short)0, acc[ns][ms], false, false);
                acc[ns][ms] = __builtin_amdgcn_wmma_f32_16x16x32_bf16(
                    false, ahi, false, blo[ms], (short)0, acc[ns][ms], false, false);
                acc[ns][ms] = __builtin_amdgcn_wmma_f32_16x16x32_bf16(
                    false, alo, false, bhi[ms], (short)0, acc[ns][ms], false, false);
            }
        }
        __syncthreads();
    }

    // -- epilogue: bias (+ReLU, + batch-stats atomics), store --------------
    // D layout: lane%16 -> WMMA-N (= m), VGPR r -> WMMA-M (= n), lanes>=16 at n+8.
    const int mloc  = lane & 15;
    const int nhalf = (lane >> 4) * 8;
    const float* bptr = bl + (long long)c * bChan;
    float* oc = Out + (long long)c * oChan;

#pragma unroll
    for (int ms = 0; ms < 2; ++ms) {
        const int mg   = bm * 128 + wm * 32 + ms * 16 + mloc;
        const float bv = bptr[mg];
        float s = 0.f, q = 0.f;
#pragma unroll
        for (int ns = 0; ns < 4; ++ns) {
            const int ng = bn * 128 + wn * 64 + ns * 16 + nhalf;
#pragma unroll
            for (int r = 0; r < 8; ++r) {
                float v = acc[ns][ms][r] + bv;
                if (doRelu) v = fmaxf(v, 0.f);
                oc[(long long)(ng + r) * oRow + mg] = v;
                s += v;
                q += v * v;
            }
        }
        if (sSum != nullptr) {
            atomicAdd(&sSum[c * L_FEAT + mg], s);
            atomicAdd(&sSq[c * L_FEAT + mg], q);
        }
    }
}

// ---------------------------------------------------------------------------
// Turn (sum, sumsq, gamma, beta) into per-(c,l) affine scale/shift:
//   scale = gamma * rsqrt(var + eps), shift = beta - mean * scale
// ---------------------------------------------------------------------------
__global__ __launch_bounds__(256)
void bn_finalize(const float* __restrict__ sSum, const float* __restrict__ sSq,
                 const float* __restrict__ gamma, const float* __restrict__ beta,
                 int gOff, float* __restrict__ scaleO, float* __restrict__ shiftO) {
    const int i = blockIdx.x * 256 + threadIdx.x;   // 0 .. C*L-1
    const int c = i >> 10;
    const int m = i & (L_FEAT - 1);
    const float mean = sSum[i] * (1.0f / N_BATCH);
    const float var  = sSq[i] * (1.0f / N_BATCH) - mean * mean;   // biased var
    const float g  = gamma[(c * 2 + gOff) * L_FEAT + m];
    const float bb = beta [(c * 2 + gOff) * L_FEAT + m];
    const float sc = g * rsqrtf(var + BN_EPS);
    scaleO[i] = sc;
    shiftO[i] = bb - mean * sc;
}

// ---------------------------------------------------------------------------
extern "C" void kernel_launch(void* const* d_in, const int* in_sizes, int n_in,
                              void* d_out, int out_size, void* d_ws, size_t ws_size,
                              hipStream_t stream) {
    const float* x     = (const float*)d_in[0];   // (N, C, L)
    const float* W     = (const float*)d_in[1];   // (C, 3, L, L)
    const float* b     = (const float*)d_in[2];   // (C, 3, L)
    const float* gamma = (const float*)d_in[3];   // (C, 2, L)
    const float* beta  = (const float*)d_in[4];   // (C, 2, L)
    float* out = (float*)d_out;                   // (N, C, L)

    const size_t HN = (size_t)N_CHAN * N_BATCH * L_FEAT;   // 16M floats
    const size_t CL = (size_t)N_CHAN * L_FEAT;             // 32K floats
    float* h1    = (float*)d_ws;       // (C, N, L)
    float* h2    = h1 + HN;            // (C, N, L)
    float* sSum  = h2 + HN;            // (C, L)
    float* sSq   = sSum + CL;
    float* scale = sSq + CL;
    float* shift = scale + CL;

    const dim3 grid(L_FEAT / 128, N_BATCH / 128, N_CHAN);  // (8, 4, 32)
    const dim3 blk(256);
    const int  fzBlocks = (int)(CL / 256);                 // 128
    const size_t statsBytes = 2 * CL * sizeof(float);

    const long long WCH = 3LL * L_FEAT * L_FEAT;
    const long long BCH = 3LL * L_FEAT;
    const long long HCH = (long long)N_BATCH * L_FEAT;

    // ---- layer 0: x (N,C,L) -> h1 (C,N,L), ReLU + stats ------------------
    hipMemsetAsync(sSum, 0, statsBytes, stream);
    gemm_bf16x3<<<grid, blk, 0, stream>>>(
        x, (long long)L_FEAT, N_CHAN * L_FEAT, nullptr, nullptr,
        W + 0LL * L_FEAT * L_FEAT, WCH, b + 0 * L_FEAT, BCH,
        h1, HCH, L_FEAT, 1, sSum, sSq);
    bn_finalize<<<fzBlocks, blk, 0, stream>>>(sSum, sSq, gamma, beta, 0, scale, shift);

    // ---- layer 1: BN(h1) -> h2, ReLU + stats -----------------------------
    hipMemsetAsync(sSum, 0, statsBytes, stream);
    gemm_bf16x3<<<grid, blk, 0, stream>>>(
        h1, HCH, L_FEAT, scale, shift,
        W + 1LL * L_FEAT * L_FEAT, WCH, b + 1 * L_FEAT, BCH,
        h2, HCH, L_FEAT, 1, sSum, sSq);
    bn_finalize<<<fzBlocks, blk, 0, stream>>>(sSum, sSq, gamma, beta, 1, scale, shift);

    // ---- layer 2: BN(h2) -> out (N,C,L), plain linear --------------------
    gemm_bf16x3<<<grid, blk, 0, stream>>>(
        h2, HCH, L_FEAT, scale, shift,
        W + 2LL * L_FEAT * L_FEAT, WCH, b + 2 * L_FEAT, BCH,
        out, (long long)L_FEAT, N_CHAN * L_FEAT, 0, nullptr, nullptr);
}
